// LDAttentionBlock_14370960572449
// MI455X (gfx1250) — matrix operand order repaired and checked by reading, not compiled, not run on hardware
//
#include <hip/hip_runtime.h>
#include <hip/hip_bf16.h>
#include <math.h>

// ---------------------------------------------------------------------------
// CDNA5 (gfx1250) wave32 WMMA helpers: D = A(16x32 f16) x B(32x16 f16) + C(f32)
// Fragment layouts: A and B both use contiguous k = 16*(lane/16) + element order.
// ---------------------------------------------------------------------------
typedef __attribute__((ext_vector_type(16))) _Float16 v16h;
typedef __attribute__((ext_vector_type(4)))  _Float16 v4h;
typedef __attribute__((ext_vector_type(8)))  float    v8f;

#define WMMA(a, b, c) \
  __builtin_amdgcn_wmma_f32_16x16x32_f16(false, (a), false, (b), (short)0, (c), false, false)

// A fragment: 16x32 (MxK) from row-major f16 tile, row stride = ld halfwords.
// lane<16: M=lane, K=0..15; lane>=16: M=lane-16, K=16..31.
// Per-lane reads are two contiguous 16B chunks -> ds_load_b128.
static __device__ __forceinline__ v16h loadA(const _Float16* p, int ld) {
  int lane = threadIdx.x & 31;
  int m = lane & 15;
  int kb = (lane >> 4) << 4;            // 0 or 16
  const _Float16* r = p + m * ld;
  v16h a;
#pragma unroll
  for (int j = 0; j < 8; ++j) {
    int k = kb + (j << 1);
    a[2 * j]     = r[k];
    a[2 * j + 1] = r[k + 1];
  }
  return a;
}

// B fragment from pre-swizzled weight tile: tile = 32 lanes x 16 halves contiguous.
// One aligned 32-byte vector load per lane.
static __device__ __forceinline__ v16h loadBswz(const _Float16* p) {
  int lane = threadIdx.x & 31;
  return *((const v16h*)p + lane);
}

// B fragment where logical B[k][n] = stored[n*ld + k]  (i.e. B = stored^T).
// Per-lane contiguous 32B read -> ds_load_b128. Used for K^T and V^T tiles.
static __device__ __forceinline__ v16h loadBt(const _Float16* p, int ld) {
  int lane = threadIdx.x & 31;
  int n = lane & 15;
  int kb = (lane >> 4) << 4;
  const _Float16* r = p + n * ld;
  v16h b;
#pragma unroll
  for (int j = 0; j < 16; ++j) b[j] = r[kb + j];
  return b;
}

// float4 -> 4 packed f16 LDS store (ds_store_b64)
static __device__ __forceinline__ void st4h(_Float16* dst, float4 v) {
  v4h h;
  h[0] = (_Float16)v.x; h[1] = (_Float16)v.y;
  h[2] = (_Float16)v.z; h[3] = (_Float16)v.w;
  *(v4h*)dst = h;
}

// ---------------------------------------------------------------------------
// Weight swizzle: orig W[N][K] row-major (torch Linear [out][in]) -> f16 tiles in
// exact WMMA B-fragment order: tile (kt, nt) = 32 lanes x 16 halves contiguous,
// element (lane, idx) = W[nt*16 + (lane&15)][kt*32 + (lane>>4)*16 + idx].
// ---------------------------------------------------------------------------
__global__ void k_wswz(const float* __restrict__ src, _Float16* __restrict__ dst,
                       int K, int N) {
  int i = blockIdx.x * blockDim.x + threadIdx.x;
  if (i >= K * N) return;
  int idx  = i & 15;
  int lane = (i >> 4) & 31;
  int tile = i >> 9;
  int ntiles = N >> 4;
  int nt = tile % ntiles, kt = tile / ntiles;
  int k = kt * 32 + ((lane >> 4) << 4) + idx;
  int n = nt * 16 + (lane & 15);
  dst[i] = (_Float16)src[(long)n * K + k];
}

// ---------------------------------------------------------------------------
// Conv2d(C=128,C=128,k=2,s=2) as GEMM, output in window-partition layout
// out[((b*256+g)*16+l)*128 + c].  Block = one (b,g) window: M=16, K=512, N=128.
// ---------------------------------------------------------------------------
__global__ void k_conv(const float* __restrict__ x, const _Float16* __restrict__ Wc,
                       const float* __restrict__ bias, float* __restrict__ out) {
  __shared__ _Float16 sA[16][512];
  int blk = blockIdx.x;
  int b = blk >> 8, g = blk & 255;
  int yg = g >> 4, xg = g & 15;
  int tid = threadIdx.x;
  // stage: kw-pairs are contiguous in NCHW -> float2 loads, b32 LDS stores
  for (int idx = tid * 2; idx < 16 * 512; idx += 512) {
    int l = idx >> 9, k = idx & 511;                 // k even
    int ci = k >> 2, kh = (k >> 1) & 1;
    int yl = l >> 2, xl = l & 3;
    int iy = ((yg << 2) + yl) * 2 + kh;
    int ix = ((xg << 2) + xl) * 2;
    const float* px = x + (((long)b * 128 + ci) * 128 + iy) * 128 + ix;
    float2 v = *(const float2*)px;
    sA[l][k]     = (_Float16)v.x;
    sA[l][k + 1] = (_Float16)v.y;
  }
  __syncthreads();
  int wave = tid >> 5, lane = tid & 31;
  int n = lane & 15, mb = (lane >> 4) << 3;
  v8f acc = {};
#pragma unroll
  for (int kt = 0; kt < 16; ++kt) {                  // N=128 -> 8 ntiles
    v16h a  = loadA(&sA[0][kt * 32], 512);
    v16h bf = loadBswz(Wc + (kt * 8 + wave) * 512);
    acc = WMMA(a, bf, acc);
  }
  int cb = wave * 16;
#pragma unroll
  for (int r = 0; r < 8; ++r) {
    int l = mb + r;
    out[((long)blk * 16 + l) * 128 + cb + n] = acc[r] + bias[cb + n];
  }
}

// ---------------------------------------------------------------------------
// LocalMSA (window L=16, 2 heads, hd=64) fused: QKV WMMA + softmax + AV + residual.
// Block = one (b,g) window, 256 threads (8 waves).
// ---------------------------------------------------------------------------
__global__ void k_msa1(const float* __restrict__ in, const _Float16* __restrict__ Wq,
                       const float* __restrict__ qb, float* __restrict__ outp) {
  __shared__ _Float16 sX[16][128];
  __shared__ _Float16 sQKV[16][384];
  __shared__ float    sS[2][16][16];
  __shared__ _Float16 sP[2][16][16];
  int blk = blockIdx.x;
  const float* xin = in + (long)blk * 2048;
  int tid = threadIdx.x;
  for (int i = tid * 4; i < 2048; i += 1024)
    st4h(&sX[0][0] + i, *(const float4*)(xin + i));
  __syncthreads();
  int wave = tid >> 5, lane = tid & 31;
  int n = lane & 15, mb = (lane >> 4) << 3;
  // QKV: M=16,K=128,N=384 -> 24 N-tiles, 3 per wave
  for (int t = wave * 3; t < wave * 3 + 3; ++t) {
    v8f acc = {};
#pragma unroll
    for (int kt = 0; kt < 4; ++kt) {
      v16h a  = loadA(&sX[0][kt * 32], 128);
      v16h bf = loadBswz(Wq + (kt * 24 + t) * 512);
      acc = WMMA(a, bf, acc);
    }
#pragma unroll
    for (int r = 0; r < 8; ++r)
      sQKV[mb + r][t * 16 + n] = (_Float16)(acc[r] + qb[t * 16 + n]);
  }
  __syncthreads();
  // scores S = Q K^T / sqrt(128): wave 0/1 -> head 0/1
  if (wave < 2) {
    int h = wave, off = h * 64;
    v8f s = {};
#pragma unroll
    for (int kt = 0; kt < 2; ++kt) {
      v16h a  = loadA(&sQKV[0][off + kt * 32], 384);
      v16h bf = loadBt(&sQKV[0][128 + off + kt * 32], 384);
      s = WMMA(a, bf, s);
    }
#pragma unroll
    for (int r = 0; r < 8; ++r)
      sS[h][mb + r][n] = s[r] * 0.08838834764831845f;
  }
  __syncthreads();
  if (tid < 32) {  // softmax: one row per thread (2 heads x 16 rows)
    int h = tid >> 4, m = tid & 15;
    float mx = -1e30f;
    for (int j = 0; j < 16; ++j) mx = fmaxf(mx, sS[h][m][j]);
    float e[16], sum = 0.f;
    for (int j = 0; j < 16; ++j) { e[j] = expf(sS[h][m][j] - mx); sum += e[j]; }
    float inv = 1.f / sum;
    for (int j = 0; j < 16; ++j) sP[h][m][j] = (_Float16)(e[j] * inv);
  }
  __syncthreads();
  // O = P V (tiny: 2*16*64) + residual
  for (int idx = tid; idx < 2048; idx += 256) {
    int h = idx >> 10, m = (idx >> 6) & 15, d = idx & 63;
    float o = 0.f;
#pragma unroll
    for (int j = 0; j < 16; ++j)
      o += (float)sP[h][m][j] * (float)sQKV[j][256 + h * 64 + d];
    int c = h * 64 + d;
    outp[(long)blk * 2048 + m * 128 + c] = xin[m * 128 + c] + o;
  }
}

// ---------------------------------------------------------------------------
// MLP: y = GELU(x W1^T + b1) W2^T + b2 + x.  Block = 16 tokens, 8 waves.
// nhwc_out=1 -> write window-reversed pixel layout [b][y*64+x][c] for the LSTM.
// ---------------------------------------------------------------------------
__global__ void k_mlp(const float* __restrict__ in, const _Float16* __restrict__ W1,
                      const float* __restrict__ b1, const _Float16* __restrict__ W2,
                      const float* __restrict__ b2, float* __restrict__ outp,
                      int nhwc_out) {
  __shared__ _Float16 sX[16][128];
  __shared__ _Float16 sH[16][512];
  int blk = blockIdx.x;
  const float* xin = in + (long)blk * 2048;
  int tid = threadIdx.x;
  for (int i = tid * 4; i < 2048; i += 1024)
    st4h(&sX[0][0] + i, *(const float4*)(xin + i));
  __syncthreads();
  int wave = tid >> 5, lane = tid & 31;
  int n = lane & 15, mb = (lane >> 4) << 3;
  for (int t = wave * 4; t < wave * 4 + 4; ++t) {    // hidden: 32 N-tiles
    v8f acc = {};
#pragma unroll
    for (int kt = 0; kt < 4; ++kt) {
      v16h a  = loadA(&sX[0][kt * 32], 128);
      v16h bf = loadBswz(W1 + (kt * 32 + t) * 512);
      acc = WMMA(a, bf, acc);
    }
#pragma unroll
    for (int r = 0; r < 8; ++r) {
      float v = acc[r] + b1[t * 16 + n];
      v = 0.5f * v * (1.f + erff(v * 0.7071067811865475f));   // exact GELU
      sH[mb + r][t * 16 + n] = (_Float16)v;
    }
  }
  __syncthreads();
  v8f acc = {};
#pragma unroll
  for (int kt = 0; kt < 16; ++kt) {                  // N=128 -> 8 ntiles
    v16h a  = loadA(&sH[0][kt * 32], 512);
    v16h bf = loadBswz(W2 + (kt * 8 + wave) * 512);
    acc = WMMA(a, bf, acc);
  }
  int cb = wave * 16;
#pragma unroll
  for (int r = 0; r < 8; ++r) {
    int m = mb + r, c = cb + n;
    float y = acc[r] + b2[c] + xin[m * 128 + c];
    if (!nhwc_out) {
      outp[(long)blk * 2048 + m * 128 + c] = y;
    } else {
      int b = blk >> 8, g = blk & 255;
      int py = ((g >> 4) << 2) + (m >> 2);
      int px = ((g & 15) << 2) + (m & 3);
      outp[((long)b * 4096 + py * 64 + px) * 128 + c] = y;
    }
  }
}

// ---------------------------------------------------------------------------
// DilatedMSA: attend over 256 windows at fixed token position l.
// Block = (b, l, head): 128 threads (4 waves), 192 KB dynamic LDS (CDNA5 320KB/WGP).
// V is stored transposed [64][256] so P*V B-fragments are contiguous loads.
// ---------------------------------------------------------------------------
__global__ void k_msa2(const float* __restrict__ in, const _Float16* __restrict__ Wq,
                       const float* __restrict__ qb, float* __restrict__ outp) {
  extern __shared__ char smem_raw[];
  _Float16* sX  = (_Float16*)smem_raw;                 // [256][128] 64KB (score scratch later)
  _Float16* sQ  = (_Float16*)(smem_raw + 65536);       // [256][64]  32KB
  _Float16* sK  = sQ + 256 * 64;                       // [256][64]  32KB
  _Float16* sVt = sK + 256 * 64;                       // [64][256]  32KB (transposed)
  int blk = blockIdx.x;
  int h = blk & 1, l = (blk >> 1) & 15, b = blk >> 5;
  int tid = threadIdx.x, wave = tid >> 5, lane = tid & 31;
  int n = lane & 15, mb = (lane >> 4) << 3;
  const float* base = in + (long)b * 524288 + (long)l * 128;  // token g stride = 2048
  for (int i = tid * 4; i < 256 * 128; i += 512) {
    int g = i >> 7, c = i & 127;
    st4h(sX + i, *(const float4*)(base + (long)g * 2048 + c));
  }
  __syncthreads();
  // QKV for this head: M=256, K=128, N=3*64 -> 16 M-tiles x 12 N-tiles
  for (int t = wave; t < 192; t += 4) {
    int mt = t / 12, nt = t % 12;
    int section = nt >> 2, d0 = (nt & 3) << 4;
    v8f acc = {};
#pragma unroll
    for (int kt = 0; kt < 4; ++kt) {
      v16h a  = loadA(sX + mt * 16 * 128 + kt * 32, 128);
      v16h bf = loadBswz(Wq + (kt * 24 + section * 8 + h * 4 + (nt & 3)) * 512);
      acc = WMMA(a, bf, acc);
    }
    int col = section * 128 + h * 64 + d0 + n;
    if (section == 2) {
#pragma unroll
      for (int r = 0; r < 8; ++r)
        sVt[(d0 + n) * 256 + mt * 16 + mb + r] = (_Float16)(acc[r] + qb[col]);
    } else {
      _Float16* dst = section ? sK : sQ;
#pragma unroll
      for (int r = 0; r < 8; ++r)
        dst[(mt * 16 + mb + r) * 64 + d0 + n] = (_Float16)(acc[r] + qb[col]);
    }
  }
  __syncthreads();   // X region now dead -> per-wave score scratch
  float*    sS  = (float*)(smem_raw + wave * 16384);             // [16][256] f32
  _Float16* sPw = (_Float16*)(smem_raw + 163840 + wave * 8192);  // [16][256] f16
  for (int mt = wave; mt < 16; mt += 4) {
    for (int ntile = 0; ntile < 16; ++ntile) {
      v8f s = {};
#pragma unroll
      for (int kt = 0; kt < 2; ++kt) {
        v16h a  = loadA(sQ + mt * 16 * 64 + kt * 32, 64);
        v16h bf = loadBt(sK + ntile * 16 * 64 + kt * 32, 64);
        s = WMMA(a, bf, s);
      }
#pragma unroll
      for (int r = 0; r < 8; ++r)
        sS[(mb + r) * 256 + ntile * 16 + n] = s[r] * 0.08838834764831845f;
    }
    __syncthreads();
    if (lane < 16) {   // softmax over 256 keys, one query row per lane
      float* row = sS + lane * 256;
      float mx = -1e30f;
      for (int j = 0; j < 256; ++j) mx = fmaxf(mx, row[j]);
      float sum = 0.f;
      for (int j = 0; j < 256; ++j) { row[j] = expf(row[j] - mx); sum += row[j]; }
      float inv = 1.f / sum;
      _Float16* pr = sPw + lane * 256;
      for (int j = 0; j < 256; ++j) pr[j] = (_Float16)(row[j] * inv);
    }
    __syncthreads();
    // O = P V : M=16, K=256, N=64 (+ residual); V^T tiles are contiguous reads
    for (int nt2 = 0; nt2 < 4; ++nt2) {
      v8f acc = {};
#pragma unroll
      for (int kt = 0; kt < 8; ++kt) {
        v16h a  = loadA(sPw + kt * 32, 256);
        v16h bf = loadBt(sVt + (nt2 * 16) * 256 + kt * 32, 256);
        acc = WMMA(a, bf, acc);
      }
      int d0 = nt2 * 16;
#pragma unroll
      for (int r = 0; r < 8; ++r) {
        int g = mt * 16 + mb + r, c = h * 64 + d0 + n;
        float y = acc[r] + base[(long)g * 2048 + c];
        outp[(long)b * 524288 + (long)g * 2048 + l * 128 + c] = y;
      }
    }
    __syncthreads();
  }
}

// ---------------------------------------------------------------------------
// State init: NCHW [128][4096] -> pixel-major [4096][128]
// ---------------------------------------------------------------------------
__global__ void k_init(const float* __restrict__ h0, const float* __restrict__ c0,
                       float* __restrict__ hs_, float* __restrict__ cs_) {
  int i = blockIdx.x * blockDim.x + threadIdx.x;
  if (i < 128 * 4096) {
    int c = i >> 12, p = i & 4095;
    hs_[(long)p * 128 + c] = h0[i];
    cs_[(long)p * 128 + c] = c0[i];
  }
}

// ---------------------------------------------------------------------------
// ConvLSTM step: gates = x Wx^T + h Wh^T + b (WMMA), channel-softmax f/i/o,
// Cn = f*C + i*tanh(s), hn = o*tanh(Cn).  Block = 16 pixels, 8 waves.
// ---------------------------------------------------------------------------
__global__ void k_lstm(const float* __restrict__ xin, const _Float16* __restrict__ Wx,
                       const _Float16* __restrict__ Wh, const float* __restrict__ bh,
                       float* __restrict__ hstate, float* __restrict__ cstate,
                       float* __restrict__ out_hs, float* __restrict__ out_hf,
                       float* __restrict__ out_cf, int last) {
  __shared__ _Float16 sX[16][128];
  __shared__ _Float16 sH[16][128];
  __shared__ float    sG[16][512];
  int pix0 = blockIdx.x * 16;
  int tid = threadIdx.x;
  for (int i = tid * 4; i < 2048; i += 1024) {
    st4h(&sX[0][0] + i, *(const float4*)(xin + (long)pix0 * 128 + i));
    st4h(&sH[0][0] + i, *(const float4*)(hstate + (long)pix0 * 128 + i));
  }
  __syncthreads();
  int wave = tid >> 5, lane = tid & 31;
  int n = lane & 15, mb = (lane >> 4) << 3;
  for (int t = wave * 4; t < wave * 4 + 4; ++t) {    // 32 N-tiles of 512 gates
    v8f acc = {};
#pragma unroll
    for (int kt = 0; kt < 4; ++kt) {
      v16h ax = loadA(&sX[0][kt * 32], 128);
      v16h bx = loadBswz(Wx + (kt * 32 + t) * 512);
      acc = WMMA(ax, bx, acc);
      v16h ah = loadA(&sH[0][kt * 32], 128);
      v16h bw = loadBswz(Wh + (kt * 32 + t) * 512);
      acc = WMMA(ah, bw, acc);
    }
#pragma unroll
    for (int r = 0; r < 8; ++r) sG[mb + r][t * 16 + n] = acc[r] + bh[t * 16 + n];
  }
  __syncthreads();
  // wave handles 2 pixels; 32 lanes x 4 channels; wave32 shuffle reductions
  for (int pp = 0; pp < 2; ++pp) {
    int p = wave * 2 + pp;
    const float* G = sG[p];
    float f[4], iv[4], sv[4], ov[4];
    float fm = -1e30f, im = -1e30f, om = -1e30f;
#pragma unroll
    for (int j = 0; j < 4; ++j) {
      int c = lane * 4 + j;
      f[j] = G[c]; iv[j] = G[128 + c]; sv[j] = G[256 + c]; ov[j] = G[384 + c];
      fm = fmaxf(fm, f[j]); im = fmaxf(im, iv[j]); om = fmaxf(om, ov[j]);
    }
    for (int d = 16; d; d >>= 1) {
      fm = fmaxf(fm, __shfl_xor(fm, d, 32));
      im = fmaxf(im, __shfl_xor(im, d, 32));
      om = fmaxf(om, __shfl_xor(om, d, 32));
    }
    float fs = 0.f, is = 0.f, os = 0.f;
#pragma unroll
    for (int j = 0; j < 4; ++j) {
      f[j] = expf(f[j] - fm); iv[j] = expf(iv[j] - im); ov[j] = expf(ov[j] - om);
      fs += f[j]; is += iv[j]; os += ov[j];
    }
    for (int d = 16; d; d >>= 1) {
      fs += __shfl_xor(fs, d, 32);
      is += __shfl_xor(is, d, 32);
      os += __shfl_xor(os, d, 32);
    }
    float fi = 1.f / fs, ii = 1.f / is, oi = 1.f / os;
#pragma unroll
    for (int j = 0; j < 4; ++j) {
      int c = lane * 4 + j;
      long gp = (long)(pix0 + p) * 128 + c;
      float Cp = cstate[gp];
      float Cn = (f[j] * fi) * Cp + (iv[j] * ii) * tanhf(sv[j]);
      float hn = (ov[j] * oi) * tanhf(Cn);
      cstate[gp] = Cn;
      hstate[gp] = hn;
      out_hs[(long)c * 4096 + pix0 + p] = hn;                       // NCHW
      if (last) {
        out_hf[(long)c * 4096 + pix0 + p] = hn;
        out_cf[(long)c * 4096 + pix0 + p] = Cn;
      }
    }
  }
}

// ---------------------------------------------------------------------------
extern "C" void kernel_launch(void* const* d_in, const int* in_sizes, int n_in,
                              void* d_out, int out_size, void* d_ws, size_t ws_size,
                              hipStream_t stream) {
  (void)in_sizes; (void)n_in; (void)out_size; (void)ws_size;
  const float* x      = (const float*)d_in[0];
  const float* h0     = (const float*)d_in[1];
  const float* c0     = (const float*)d_in[2];
  const float* conv_w = (const float*)d_in[3];
  const float* conv_b = (const float*)d_in[4];
  const float* qkv1_w = (const float*)d_in[5];
  const float* qkv1_b = (const float*)d_in[6];
  const float* m1w1   = (const float*)d_in[7];
  const float* m1b1   = (const float*)d_in[8];
  const float* m1w2   = (const float*)d_in[9];
  const float* m1b2   = (const float*)d_in[10];
  const float* qkv2_w = (const float*)d_in[11];
  const float* qkv2_b = (const float*)d_in[12];
  const float* m2w1   = (const float*)d_in[13];
  const float* m2b1   = (const float*)d_in[14];
  const float* m2w2   = (const float*)d_in[15];
  const float* m2b2   = (const float*)d_in[16];
  const float* lwx    = (const float*)d_in[17];
  const float* lwh    = (const float*)d_in[18];
  const float* lbh    = (const float*)d_in[19];

  char* ws = (char*)d_ws;
  const size_t ACT = (size_t)16 * 256 * 16 * 128 * 4;   // 32 MB
  float* bufA   = (float*)ws; ws += ACT;
  float* bufB   = (float*)ws; ws += ACT;
  float* hstate = (float*)ws; ws += (size_t)4096 * 128 * 4;
  float* cstate = (float*)ws; ws += (size_t)4096 * 128 * 4;
  _Float16* Wconv = (_Float16*)ws; ws += 512 * 128 * 2;
  _Float16* Wqkv1 = (_Float16*)ws; ws += 128 * 384 * 2;
  _Float16* Wm1a  = (_Float16*)ws; ws += 128 * 512 * 2;
  _Float16* Wm1b  = (_Float16*)ws; ws += 512 * 128 * 2;
  _Float16* Wqkv2 = (_Float16*)ws; ws += 128 * 384 * 2;
  _Float16* Wm2a  = (_Float16*)ws; ws += 128 * 512 * 2;
  _Float16* Wm2b  = (_Float16*)ws; ws += 512 * 128 * 2;
  _Float16* Wlx   = (_Float16*)ws; ws += 128 * 512 * 2;
  _Float16* Wlh   = (_Float16*)ws; ws += 128 * 512 * 2;

  // weights -> f16, WMMA B-fragment swizzled (orig is [N][K] row-major)
  k_wswz<<<256, 256, 0, stream>>>(conv_w, Wconv, 512, 128);
  k_wswz<<<192, 256, 0, stream>>>(qkv1_w, Wqkv1, 128, 384);
  k_wswz<<<256, 256, 0, stream>>>(m1w1,   Wm1a,  128, 512);
  k_wswz<<<256, 256, 0, stream>>>(m1w2,   Wm1b,  512, 128);
  k_wswz<<<192, 256, 0, stream>>>(qkv2_w, Wqkv2, 128, 384);
  k_wswz<<<256, 256, 0, stream>>>(m2w1,   Wm2a,  128, 512);
  k_wswz<<<256, 256, 0, stream>>>(m2w2,   Wm2b,  512, 128);
  k_wswz<<<256, 256, 0, stream>>>(lwx,    Wlx,   128, 512);
  k_wswz<<<256, 256, 0, stream>>>(lwh,    Wlh,   128, 512);

  k_conv<<<4096, 256, 0, stream>>>(x, Wconv, conv_b, bufA);
  k_msa1<<<4096, 256, 0, stream>>>(bufA, Wqkv1, qkv1_b, bufB);
  k_mlp <<<4096, 256, 0, stream>>>(bufB, Wm1a, m1b1, Wm1b, m1b2, bufA, 0);
  k_msa2<<<512, 128, 196608, stream>>>(bufA, Wqkv2, qkv2_b, bufB);
  k_mlp <<<4096, 256, 0, stream>>>(bufB, Wm2a, m2b1, Wm2b, m2b2, bufA, 1);

  float* out = (float*)d_out;
  float* out_hf = out + (size_t)16 * 524288;
  float* out_cf = out_hf + 524288;
  k_init<<<2048, 256, 0, stream>>>(h0, c0, hstate, cstate);
  for (int step = 0; step < 16; ++step) {
    k_lstm<<<256, 256, 0, stream>>>(bufA + (size_t)step * 524288, Wlx, Wlh, lbh,
                                    hstate, cstate, out + (size_t)step * 524288,
                                    out_hf, out_cf, step == 15);
  }
}
